// Attention_14130442404175
// MI455X (gfx1250) — matrix operand order repaired and checked by reading, not compiled, not run on hardware
//
#include <hip/hip_runtime.h>

// CDNA5 / gfx1250 fused attention:
//   scores = (Q K^T)/sqrt(D), key-mask by valid_len, softmax, ctx = P V
// Outputs (concatenated): ctx (B,Q,H,D) f32, attn (B,H,Q,K) f32.
// GEMMs: v_wmma_f32_16x16x32_bf16. V staged via Tensor Data Mover (TDM)
// double-buffered into padded LDS, synced with s_wait_tensorcnt.
// Normalized P cached in LDS as bf16 so the P.V loop needs no converts.

typedef __attribute__((ext_vector_type(16))) __bf16 v16bf;
typedef __attribute__((ext_vector_type(8)))  __bf16 v8bf;
typedef __attribute__((ext_vector_type(4)))  __bf16 v4bf;
typedef __attribute__((ext_vector_type(8)))  float  v8f;
typedef __attribute__((ext_vector_type(4)))  unsigned v4u;
typedef __attribute__((ext_vector_type(8)))  int  v8i_;
typedef __attribute__((ext_vector_type(4)))  int  v4i_;

#if defined(__has_builtin)
#  if __has_builtin(__builtin_amdgcn_tensor_load_to_lds)
#    define USE_TDM 1
#  else
#    define USE_TDM 0
#  endif
#else
#  define USE_TDM 0
#endif

namespace {
constexpr int B_ = 2;
constexpr int Q_ = 2048;
constexpr int K_ = 2048;
constexpr int H_ = 16;
constexpr int D_ = 128;
constexpr float SCALE = 0.08838834764831845f;  // 1/sqrt(128)

constexpr int WAVES   = 8;   // 256 threads
constexpr int THREADS = WAVES * 32;
constexpr int QTILES  = Q_ / 16;  // 128

constexpr int VROWS   = 64;            // V rows per TDM chunk
constexpr int NCHUNK  = K_ / VROWS;    // 32
constexpr int VPITCH  = 129;           // dwords per LDS V row (TDM pad: +1 dw / 128 dw)
constexpr int PPITCH  = K_ + 8;        // bf16 P row pitch (2056): banks spread for b128 reads

// LDS layout (byte offsets into dynamic smem)
constexpr unsigned OFF_SCORES = 0;                                    // 16 x 2048 f32
constexpr unsigned OFF_Q      = OFF_SCORES + 16u * K_ * 4u;           // 16 x 128 bf16
constexpr unsigned OFF_RED    = OFF_Q + 16u * D_ * 2u;                // 256 f32
constexpr unsigned OFF_STAT   = OFF_RED + 256u * 4u;                  // 32 f32
constexpr unsigned OFF_PB     = OFF_STAT + 32u * 4u;                  // 16 x 2056 bf16
constexpr unsigned OFF_V0     = OFF_PB + 16u * (unsigned)PPITCH * 2u; // 64 x 129 f32
constexpr unsigned OFF_V1     = OFF_V0 + (unsigned)(VROWS * VPITCH * 4);
constexpr size_t   SMEM_BYTES = OFF_V1 + (size_t)(VROWS * VPITCH * 4);
}  // namespace

__device__ __forceinline__ __bf16 f2bf(float f) { return (__bf16)f; }  // v_cvt_pk_bf16_f32

#if USE_TDM
// Issue a TDM 2D tile load: VROWS x 128 f32, row stride H_*D_ elements,
// into LDS at lds_off with 1-dword padding per 128-dword row (bank-conflict kill).
__device__ __forceinline__ void tdm_load_v(unsigned lds_off, const float* gptr) {
  const unsigned long long ga = (unsigned long long)(__SIZE_TYPE__)gptr;
  v4u g0;
  g0[0] = 1u;                                            // count=1, user descriptor
  g0[1] = lds_off;                                       // lds_addr (bytes)
  g0[2] = (unsigned)(ga & 0xFFFFFFFFu);                  // global_addr[31:0]
  g0[3] = (unsigned)((ga >> 32) & 0x1FFFFFFu) | (2u << 30);  // ga[56:32] | type=2
  v8i_ g1;
  g1[0] = (int)((2u << 16) | (1u << 20) | (6u << 22));   // data_size=4B, pad_en, ivl=128dw, amt=1dw
  g1[1] = (int)(128u << 16);                             // tensor_dim0 = 128
  g1[2] = (int)((unsigned)K_ << 16);                     // tensor_dim1 = 2048
  g1[3] = (int)(128u << 16);                             // tile_dim0 = 128
  g1[4] = VROWS;                                         // tile_dim1 = 64 (tile_dim2=0 -> 2D)
  g1[5] = H_ * D_;                                       // tensor_dim0_stride = 2048
  g1[6] = 0;
  g1[7] = 0;
  v4i_ z = {0, 0, 0, 0};
#if defined(__clang_major__) && __clang_major__ >= 23
  v8i_ z8 = {0, 0, 0, 0, 0, 0, 0, 0};
  __builtin_amdgcn_tensor_load_to_lds(g0, g1, z, z, z8, 0);
#else
  __builtin_amdgcn_tensor_load_to_lds(g0, g1, z, z, 0);
#endif
}
#endif  // USE_TDM

__global__ __launch_bounds__(THREADS)
void Attention_14130442404175_kernel(const float* __restrict__ qs,
                                     const float* __restrict__ ks,
                                     const float* __restrict__ vs,
                                     const int*   __restrict__ valid_lens,
                                     float* __restrict__ ctx_out,
                                     float* __restrict__ attn_out) {
  extern __shared__ char smem[];
  float*  sScores = reinterpret_cast<float*>(smem + OFF_SCORES);  // [16][K_]
  __bf16* sQ      = reinterpret_cast<__bf16*>(smem + OFF_Q);      // [16][D_]
  float*  sRed    = reinterpret_cast<float*>(smem + OFF_RED);
  float*  sRowMax = reinterpret_cast<float*>(smem + OFF_STAT);
  float*  sRowInv = sRowMax + 16;
  __bf16* sPb     = reinterpret_cast<__bf16*>(smem + OFF_PB);     // [16][PPITCH]
  float*  sV0     = reinterpret_cast<float*>(smem + OFF_V0);
  float*  sV1     = reinterpret_cast<float*>(smem + OFF_V1);

  const int tid  = threadIdx.x;
  const int lane = tid & 31;
  const int wave = tid >> 5;
  const int hi   = (lane >> 4) & 1;
  const int lo   = lane & 15;

  const int bh = blockIdx.x / QTILES;
  const int qt = blockIdx.x % QTILES;
  const int b  = bh / H_;
  const int h  = bh % H_;
  const int q0 = qt * 16;
  const int vlen = valid_lens[b];

  // ---- Stage Q tile (16 x 128) into LDS as bf16 (coalesced) ----
  for (int idx = tid; idx < 16 * D_; idx += THREADS) {
    const int r = idx >> 7;
    const int d = idx & (D_ - 1);
    sQ[r * D_ + d] = f2bf(qs[((size_t)(b * Q_ + q0 + r) * H_ + h) * D_ + d]);
  }
  __syncthreads();

  // ---- Hoisted Q A-fragments (loop-invariant over k-tiles) ----
  // A (16x32 bf16): lane row m = lo; elems 0..7 <- k = hi*8+0..7, 8..15 <- k = 16+hi*8+0..7
  v16bf aQ[4];
  {
    const __bf16* qrow = sQ + lo * D_;
#pragma unroll
    for (int dc = 0; dc < 4; ++dc) {
#pragma unroll
      for (int t = 0; t < 8; ++t) aQ[dc][t] = qrow[dc * 32 + hi * 8 + t];
#pragma unroll
      for (int t = 0; t < 8; ++t) aQ[dc][8 + t] = qrow[dc * 32 + 16 + hi * 8 + t];
    }
  }

  // ---- Phase 1: score strip (16 x 2048) into LDS via WMMA ----
  // B (32x16 bf16): lane col n = lo; elems 0..7 <- k = hi*8+0..7, 8..15 <- k = 16+hi*8+0..7
  // C/D (16x16 f32): vgpr r holds row m = r + hi*8, col n = lo
  for (int kt = wave; kt < K_ / 16; kt += WAVES) {
    v8f acc = {};
    const float* krow = ks + ((size_t)(b * K_ + kt * 16 + lo) * H_ + h) * D_;
#pragma unroll
    for (int dc = 0; dc < D_ / 32; ++dc) {
      v16bf bf;
      const float* kk = krow + dc * 32 + hi * 8;  // two runs of 8 f32 (k = hi*8+t, 16+hi*8+t)
#pragma unroll
      for (int t = 0; t < 16; ++t) bf[t] = f2bf(kk[t + (t & 8)]);
      acc = __builtin_amdgcn_wmma_f32_16x16x32_bf16(
          false, aQ[dc], false, bf, (short)0, acc, false, false);
    }
    const int kcol = kt * 16 + lo;
    const bool ok  = kcol < vlen;
#pragma unroll
    for (int r = 0; r < 8; ++r) {
      const int m = r + hi * 8;
      sScores[m * K_ + kcol] = ok ? acc[r] * SCALE : -1.0e6f;
    }
  }
  __syncthreads();

  // ---- Phase 2: row softmax over the LDS strip (float4-vectorized) ----
  {
    const int row = tid >> 4;  // 0..15
    const int seg = tid & 15;  // 128 contiguous cols per thread
    float4* rp4 = reinterpret_cast<float4*>(sScores + row * K_ + seg * 128);
    float mx = -3.4e38f;
    for (int i = 0; i < 32; ++i) {
      const float4 v = rp4[i];
      mx = fmaxf(mx, fmaxf(fmaxf(v.x, v.y), fmaxf(v.z, v.w)));
    }
    sRed[tid] = mx;
    __syncthreads();
    if (tid < 16) {
      float m2 = sRed[tid * 16];
      for (int i = 1; i < 16; ++i) m2 = fmaxf(m2, sRed[tid * 16 + i]);
      sRowMax[tid] = m2;
    }
    __syncthreads();
    const float rmax = sRowMax[row];
    float sum = 0.0f;
    for (int i = 0; i < 32; ++i) {
      float4 v = rp4[i];
      v.x = __expf(v.x - rmax); v.y = __expf(v.y - rmax);
      v.z = __expf(v.z - rmax); v.w = __expf(v.w - rmax);
      rp4[i] = v;
      sum += (v.x + v.y) + (v.z + v.w);
    }
    sRed[tid] = sum;
    __syncthreads();
    if (tid < 16) {
      float s2 = 0.0f;
      for (int i = 0; i < 16; ++i) s2 += sRed[tid * 16 + i];
      sRowInv[tid] = 1.0f / s2;
    }
    __syncthreads();
  }

  // ---- Normalize: one coalesced HBM write of attn (b128) + bf16 P into LDS ----
  float4* attn4 = reinterpret_cast<float4*>(attn_out + ((size_t)(b * H_ + h) * Q_ + q0) * K_);
  const float4* sc4 = reinterpret_cast<const float4*>(sScores);
  for (int idx = tid; idx < 16 * (K_ / 4); idx += THREADS) {
    const int r  = idx >> 9;        // K_/4 == 512
    const int c4 = idx & 511;
    const float inv = sRowInv[r];
    float4 v = sc4[idx];
    v.x *= inv; v.y *= inv; v.z *= inv; v.w *= inv;
    attn4[idx] = v;
    v4bf p;  // packed v_cvt_pk_bf16_f32 x2, 8B ds_store
    p[0] = f2bf(v.x); p[1] = f2bf(v.y); p[2] = f2bf(v.z); p[3] = f2bf(v.w);
    *reinterpret_cast<v4bf*>(sPb + r * PPITCH + c4 * 4) = p;
  }
  __syncthreads();

  // ---- Phase 3: ctx = P (16 x 2048) x V (2048 x 128); one 16-col D tile/wave ----
  // V chunks (64 x 128 f32) staged into padded LDS (pitch 129 dw) via TDM,
  // double-buffered, DMA overlapped with WMMA. P read as bf16 b128 LDS loads.
  {
    const int dtile = wave;
    const float* vbase = vs + ((size_t)(b * K_) * H_ + h) * D_;
    const __bf16* prow = sPb + lo * PPITCH;
    v8f acc = {};
#if USE_TDM
    if (wave == 0) {
      tdm_load_v(OFF_V0, vbase);
      __builtin_amdgcn_s_wait_tensorcnt(0);
    }
    __syncthreads();
    for (int kc = 0; kc < NCHUNK; ++kc) {
      const int cur = kc & 1;
      if (wave == 0 && kc + 1 < NCHUNK)
        tdm_load_v(cur ? OFF_V0 : OFF_V1,
                   vbase + (size_t)(kc + 1) * VROWS * (H_ * D_));
      const float* vb = cur ? sV1 : sV0;
#pragma unroll
      for (int sub = 0; sub < 2; ++sub) {
        const int kb = kc * VROWS + sub * 32;
        const v8bf pl = *reinterpret_cast<const v8bf*>(prow + kb + hi * 8);
        const v8bf ph = *reinterpret_cast<const v8bf*>(prow + kb + 16 + hi * 8);
        const v16bf a = __builtin_shufflevector(pl, ph, 0, 1, 2, 3, 4, 5, 6, 7,
                                                8, 9, 10, 11, 12, 13, 14, 15);
        v16bf bf;
        const float* vcol = vb + (sub * 32 + hi * 8) * VPITCH + dtile * 16 + lo;
#pragma unroll
        for (int t = 0; t < 16; ++t) bf[t] = f2bf(vcol[(t + (t & 8)) * VPITCH]);
        acc = __builtin_amdgcn_wmma_f32_16x16x32_bf16(
            false, a, false, bf, (short)0, acc, false, false);
      }
      if (wave == 0 && kc + 1 < NCHUNK) __builtin_amdgcn_s_wait_tensorcnt(0);
      __syncthreads();
    }
#else
    // Fallback: cooperative staged loads (single buffer).
    for (int kc = 0; kc < NCHUNK; ++kc) {
      const float* gsrc = vbase + (size_t)kc * VROWS * (H_ * D_);
      for (int idx = tid; idx < VROWS * D_; idx += THREADS) {
        const int r = idx >> 7, c = idx & 127;
        sV0[r * VPITCH + c] = gsrc[(size_t)r * (H_ * D_) + c];
      }
      __syncthreads();
#pragma unroll
      for (int sub = 0; sub < 2; ++sub) {
        const int kb = kc * VROWS + sub * 32;
        const v8bf pl = *reinterpret_cast<const v8bf*>(prow + kb + hi * 8);
        const v8bf ph = *reinterpret_cast<const v8bf*>(prow + kb + 16 + hi * 8);
        const v16bf a = __builtin_shufflevector(pl, ph, 0, 1, 2, 3, 4, 5, 6, 7,
                                                8, 9, 10, 11, 12, 13, 14, 15);
        v16bf bf;
        const float* vcol = sV0 + (sub * 32 + hi * 8) * VPITCH + dtile * 16 + lo;
#pragma unroll
        for (int t = 0; t < 16; ++t) bf[t] = f2bf(vcol[(t + (t & 8)) * VPITCH]);
        acc = __builtin_amdgcn_wmma_f32_16x16x32_bf16(
            false, a, false, bf, (short)0, acc, false, false);
      }
      __syncthreads();
    }
#endif
#pragma unroll
    for (int r = 0; r < 8; ++r) {
      const int m = r + hi * 8;
      ctx_out[((size_t)(b * Q_ + q0 + m) * H_ + h) * D_ + dtile * 16 + lo] = acc[r];
    }
  }
}

extern "C" void kernel_launch(void* const* d_in, const int* in_sizes, int n_in,
                              void* d_out, int out_size, void* d_ws, size_t ws_size,
                              hipStream_t stream) {
  (void)in_sizes; (void)n_in; (void)d_ws; (void)ws_size; (void)out_size;
  const float* qs = (const float*)d_in[0];
  const float* ks = (const float*)d_in[1];
  const float* vs = (const float*)d_in[2];
  const int*   vl = (const int*)d_in[3];

  float* ctx  = (float*)d_out;
  float* attn = ctx + (size_t)B_ * Q_ * H_ * D_;

  (void)hipFuncSetAttribute(
      reinterpret_cast<const void*>(&Attention_14130442404175_kernel),
      hipFuncAttributeMaxDynamicSharedMemorySize, (int)SMEM_BYTES);

  const int grid = B_ * H_ * QTILES;  // 4096 workgroups
  Attention_14130442404175_kernel<<<grid, THREADS, SMEM_BYTES, stream>>>(
      qs, ks, vs, vl, ctx, attn);
}
